// HdcCNNEncoder_39359080300674
// MI455X (gfx1250) — matrix operand order repaired and checked by reading, not compiled, not run on hardware
//
#include <hip/hip_runtime.h>
#include <hip/hip_bf16.h>

typedef __attribute__((ext_vector_type(16))) _Float16 v16h;
typedef __attribute__((ext_vector_type(8)))  _Float16 v8h;
typedef __attribute__((ext_vector_type(8)))  float    v8f;
typedef __attribute__((ext_vector_type(4)))  float    v4f;

#define T_LEN    4018
#define L1       4009        // conv1 output length
#define L2       4000        // conv2 output length
#define P_LEN    2000        // pooled length
#define LIN_IN   256000      // 128 * P_LEN
#define OUT_N    10000
#define K2DIM    640         // 64 ch * 10 taps
#define CHUNK    5120        // LIN_IN / 50
#define NCHUNK   50

// ---------------------------------------------------------------------------
// K1: conv1 (4 -> 64 ch, k=10, valid) + ReLU.  64*4009 outputs, 40 MACs each.
// signals is [T,4] row-major; x[ci,t] = signals[t*4+ci].
// ---------------------------------------------------------------------------
__global__ void k_conv1(const float* __restrict__ signals,
                        const float* __restrict__ w,
                        const float* __restrict__ b,
                        float* __restrict__ relu1) {
    int gid = blockIdx.x * 256 + threadIdx.x;
    if (gid >= 64 * L1) return;
    int oc = gid / L1;
    int t  = gid - oc * L1;
    const float* wr = w + oc * 40;
    float acc = b[oc];
#pragma unroll
    for (int ci = 0; ci < 4; ++ci)
#pragma unroll
        for (int dk = 0; dk < 10; ++dk)
            acc = fmaf(signals[(t + dk) * 4 + ci], wr[ci * 10 + dk], acc);
    relu1[gid] = fmaxf(acc, 0.0f);
}

// ---------------------------------------------------------------------------
// K1b: convert conv2 weights [128][640] fp32 -> fp16 once (L2-resident after).
// ---------------------------------------------------------------------------
__global__ void k_cvt_w2(const float* __restrict__ w, _Float16* __restrict__ wh) {
    int i = blockIdx.x * 256 + threadIdx.x;
    if (i < 128 * K2DIM) wh[i] = (_Float16)w[i];
}

// ---------------------------------------------------------------------------
// K2: conv2 as WMMA GEMM  C[128,4000] = W[128,640] x X[640,4000]
//     (X = im2col of relu1), fused bias + ReLU + MaxPool(2).
// One block = 256 thr = 8 waves; block owns one N-tile of 16 columns,
// wave w owns M-tile rows [16w,16w+16).  K loop: 20 x v_wmma_f32_16x16x32_f16.
// B tile staged in LDS as f16, transposed to [n][k] so each lane's 16 K-values
// are two contiguous ds_load_b128s.  A frag = two contiguous global b128s.
// ---------------------------------------------------------------------------
__global__ __launch_bounds__(256) void k_conv2_wmma(
        const float* __restrict__ relu1, const _Float16* __restrict__ Wh,
        const float* __restrict__ conv2_b, float* __restrict__ pooled) {
    __shared__ __align__(16) _Float16 Xh[16 * K2DIM];   // 20 KB

    const int tid    = threadIdx.x;
    const int n_base = blockIdx.x * 16;

    // Build Xh[n][k] = relu1[k/10][n_base + n + k%10]  (f32 -> f16)
    for (int idx = tid; idx < 16 * K2DIM; idx += 256) {
        int n  = idx / K2DIM;
        int k  = idx - n * K2DIM;
        int c  = k / 10;
        int dk = k - c * 10;
        Xh[idx] = (_Float16)relu1[c * L1 + n_base + n + dk];
    }
    __syncthreads();

    const int lane   = tid & 31;
    const int hi     = lane >> 4;       // half-wave select
    const int nl     = lane & 15;       // N (and A-row) index within tile
    const int m_base = (tid >> 5) * 16; // wave id * 16
    const int m_row  = m_base + nl;

    const _Float16* __restrict__ arow = Wh + m_row * K2DIM;
    const _Float16* __restrict__ brow = Xh + nl * K2DIM;

    v8f acc = {};
    for (int kk = 0; kk < K2DIM; kk += 32) {
        union { v16h v; v8h h[2]; } a, bm;
        // A lanes 0-15: K = kk+{0..7, 16..23};  lanes 16-31: kk+{8..15, 24..31}
        a.h[0]  = *(const v8h*)(arow + kk + hi * 8);
        a.h[1]  = *(const v8h*)(arow + kk + hi * 8 + 16);
        // B lanes 0-15: K = kk+{0..7, 16..23};  lanes 16-31: kk+{8..15, 24..31}
        bm.h[0] = *(const v8h*)(brow + kk + hi * 8);
        bm.h[1] = *(const v8h*)(brow + kk + hi * 8 + 16);
        acc = __builtin_amdgcn_wmma_f32_16x16x32_f16(
                false, a.v, false, bm.v, (short)0, acc, false, false);
    }

    // Epilogue: C VGPR v <-> (M = m_base + v + 8*hi, N = n_base + nl).
    // bias + ReLU, then MaxPool(2) across adjacent-N lanes via shfl_xor(1).
#pragma unroll
    for (int v = 0; v < 8; ++v) {
        int   m   = m_base + v + hi * 8;
        float val = fmaxf(acc[v] + conv2_b[m], 0.0f);
        float nbv = __shfl_xor(val, 1, 32);
        float mx  = fmaxf(val, nbv);
        if ((nl & 1) == 0)
            pooled[m * P_LEN + ((n_base + nl) >> 1)] = mx;
    }
}

// ---------------------------------------------------------------------------
// K3: linear split-K partials.  grid = (50 chunks, 128 rows), 256 thr.
// Streams 131 MB of lin_w with non-temporal float4 loads (single-use data);
// pooled chunk (20 KB) stays hot in cache across the 128 rows.
// ---------------------------------------------------------------------------
__global__ __launch_bounds__(256) void k_lin_partial(
        const float* __restrict__ lin_w, const float* __restrict__ pooled,
        float* __restrict__ partial) {
    const int r = blockIdx.y, chunk = blockIdx.x, t = threadIdx.x;
    const v4f* __restrict__ wp = (const v4f*)(lin_w + (size_t)r * LIN_IN + chunk * CHUNK);
    const v4f* __restrict__ xp = (const v4f*)(pooled + chunk * CHUNK);
    float s = 0.0f;
#pragma unroll
    for (int i = 0; i < 5; ++i) {
        v4f w = __builtin_nontemporal_load(wp + i * 256 + t);
        v4f x = xp[i * 256 + t];
        s = fmaf(w.x, x.x, s); s = fmaf(w.y, x.y, s);
        s = fmaf(w.z, x.z, s); s = fmaf(w.w, x.w, s);
    }
    __shared__ float red[256];
    red[t] = s; __syncthreads();
    for (int off = 128; off > 0; off >>= 1) {
        if (t < off) red[t] += red[t + off];
        __syncthreads();
    }
    if (t == 0) partial[r * NCHUNK + chunk] = red[0];
}

// K4: reduce partials, add bias, ReLU -> xvec[128]
__global__ void k_lin_final(const float* __restrict__ partial,
                            const float* __restrict__ lin_b,
                            float* __restrict__ xvec) {
    int r = threadIdx.x;
    float s = 0.0f;
    for (int i = 0; i < NCHUNK; ++i) s += partial[r * NCHUNK + i];
    xvec[r] = fmaxf(s + lin_b[r], 0.0f);
}

// ---------------------------------------------------------------------------
// K5: HDC tail.  Per output o: 128-dot + 6-dot + 18x 3-dots, 20 sinusoids,
// bundle/bind, hard quantize.  xvec + feat cached in LDS.
// ---------------------------------------------------------------------------
__global__ __launch_bounds__(128) void k_hdc(
        const float* __restrict__ xvec, const float* __restrict__ feat,
        const float* __restrict__ hdc_w, const float* __restrict__ hdc_b,
        const float* __restrict__ mfcc_w, const float* __restrict__ mfcc_b,
        const float* __restrict__ feat3_w, const float* __restrict__ feat3_b,
        float* __restrict__ out) {
    __shared__ float xs[128];
    __shared__ float fs[60];
    const int t = threadIdx.x;
    xs[t] = xvec[t];
    if (t < 60) fs[t] = feat[t];
    __syncthreads();

    const int o = blockIdx.x * 128 + t;
    if (o >= OUT_N) return;

    const float* hw = hdc_w + (size_t)o * 128;
    float p = 0.0f;
#pragma unroll 8
    for (int j = 0; j < 128; ++j) p = fmaf(hw[j], xs[j], p);
    float sample = cosf(p + hdc_b[o]) * sinf(p);

    float m = 0.0f;
#pragma unroll
    for (int j = 0; j < 6; ++j) m = fmaf(mfcc_w[o * 6 + j], fs[j], m);
    float mf = cosf(m + mfcc_b[o]) * sinf(m);

    float hv[18];
#pragma unroll
    for (int s = 0; s < 18; ++s) {
        const float* w3 = feat3_w + ((size_t)s * OUT_N + o) * 3;
        float q = w3[0] * fs[6 + s * 3] + w3[1] * fs[7 + s * 3] + w3[2] * fs[8 + s * 3];
        hv[s] = cosf(q + feat3_b[s * OUT_N + o]) * sinf(q);
    }
    float g1 = hv[0] + hv[7] + hv[15] + hv[16] + hv[17];
    float g2 = hv[1] + hv[2] + hv[3] + hv[4] + hv[5];
    float g3 = mf + hv[8] + hv[9] + hv[10] + hv[11] + hv[12] + hv[13] + hv[14];
    float v  = sample * g1 * g2 * hv[6] * g3;
    out[o] = v > 0.0f ? 1.0f : -1.0f;
}

// ---------------------------------------------------------------------------
extern "C" void kernel_launch(void* const* d_in, const int* in_sizes, int n_in,
                              void* d_out, int out_size, void* d_ws, size_t ws_size,
                              hipStream_t stream) {
    const float* signals = (const float*)d_in[0];
    const float* feat    = (const float*)d_in[1];
    const float* conv1_w = (const float*)d_in[2];
    const float* conv1_b = (const float*)d_in[3];
    const float* conv2_w = (const float*)d_in[4];
    const float* conv2_b = (const float*)d_in[5];
    const float* lin_w   = (const float*)d_in[6];
    const float* lin_b   = (const float*)d_in[7];
    const float* hdc_w   = (const float*)d_in[8];
    const float* hdc_b   = (const float*)d_in[9];
    const float* mfcc_w  = (const float*)d_in[10];
    const float* mfcc_b  = (const float*)d_in[11];
    const float* feat3_w = (const float*)d_in[12];
    const float* feat3_b = (const float*)d_in[13];
    float* out = (float*)d_out;

    // workspace carve-up (~2.2 MB total, 16B-aligned offsets)
    char* ws = (char*)d_ws;
    float*     relu1   = (float*)    (ws + 0);          //  64*4009 f32 = 1,026,304 B
    _Float16*  Wh      = (_Float16*) (ws + 1026304);    // 128*640  f16 =   163,840 B
    float*     pooled  = (float*)    (ws + 1190144);    // 128*2000 f32 = 1,024,000 B
    float*     partial = (float*)    (ws + 2214144);    // 128*50   f32 =    25,600 B
    float*     xvec    = (float*)    (ws + 2239744);    // 128      f32 =       512 B

    // conv1 + ReLU
    k_conv1<<<(64 * L1 + 255) / 256, 256, 0, stream>>>(signals, conv1_w, conv1_b, relu1);
    // conv2 weights -> f16
    k_cvt_w2<<<(128 * K2DIM + 255) / 256, 256, 0, stream>>>(conv2_w, Wh);
    // conv2 WMMA GEMM + bias + ReLU + maxpool
    k_conv2_wmma<<<L2 / 16, 256, 0, stream>>>(relu1, Wh, conv2_b, pooled);
    // linear: streaming split-K then reduce
    k_lin_partial<<<dim3(NCHUNK, 128), 256, 0, stream>>>(lin_w, pooled, partial);
    k_lin_final<<<1, 128, 0, stream>>>(partial, lin_b, xvec);
    // HDC sinusoid encode + bind + hard quantize
    k_hdc<<<(OUT_N + 127) / 128, 128, 0, stream>>>(xvec, feat, hdc_w, hdc_b,
                                                   mfcc_w, mfcc_b, feat3_w, feat3_b, out);
}
